// edge_CrossAttention_63324997812372
// MI455X (gfx1250) — matrix operand order repaired and checked by reading, not compiled, not run on hardware
//
#include <hip/hip_runtime.h>

// ---------------------------------------------------------------------------
// edge CrossAttention for MI455X (gfx1250, wave32, WMMA + async/TDM staging).
// Pipeline:
//   1) cast fp32 -> bf16 (q, kv, wq, wk, wv, w_proj)
//   2) qp = q·wq^T, kp = kv·wk^T, vp = kv·wv^T   (bf16 WMMA GEMM, f32 accum,
//      A tile staged via global_load_async_to_lds_b128, B tile via TDM)
//   3) vt = per-head transpose of vp (so P·V contraction is contiguous)
//   4) flash attention per (b,h,64-row Q tile): S=Q·K^T (WMMA), online
//      softmax (f32), P·V (WMMA) -> x (bf16)
//   5) out = x·w_proj^T + b_proj (WMMA, fp32 out)
// ---------------------------------------------------------------------------

typedef __attribute__((ext_vector_type(16))) __bf16 v16bf;
typedef __attribute__((ext_vector_type(8)))  __bf16 v8bf;
typedef __attribute__((ext_vector_type(8)))  float  v8f;
typedef unsigned int uint32x4 __attribute__((ext_vector_type(4)));
typedef int          int32x4  __attribute__((ext_vector_type(4)));
typedef int          int32x8  __attribute__((ext_vector_type(8)));

#define DIM       768
#define DIM_EDGE  256
#define NHEAD     8
#define HDIM      96
#define BATCH     8
#define NQ        1024
#define NK        1024
#define MROWS     (BATCH * NQ)   // 8192

#if defined(__has_builtin)
#if __has_builtin(__builtin_amdgcn_tensor_load_to_lds)
#define HAVE_TDM 1
#else
#define HAVE_TDM 0
#endif
#else
#define HAVE_TDM 0
#endif

__device__ __forceinline__ v8f wmma_bf16(v16bf a, v16bf b, v8f c) {
  return __builtin_amdgcn_wmma_f32_16x16x32_bf16(
      /*neg_a=*/false, a, /*neg_b=*/false, b,
      /*c_mod=*/(short)0, c, /*reuse_a=*/false, /*reuse_b=*/false);
}

// Build a 16x32 A-fragment (contiguous k order, same as the B loads): lane holds row
// m = lane&15, K elements {off..off+15} with off = 16*(lane>=16).
__device__ __forceinline__ v16bf load_frag2(const __bf16* p0, const __bf16* p1) {
  union { v16bf v; v8bf h[2]; } u;
  u.h[0] = *(const v8bf*)p0;
  u.h[1] = *(const v8bf*)p1;
  return u.v;
}

// ---- async copy: one 16B chunk global -> LDS, tracked by ASYNCcnt ----------
__device__ __forceinline__ void async_copy_b128(unsigned int lds_off,
                                                const void* gsrc) {
  asm volatile("global_load_async_to_lds_b128 %0, %1, off"
               :: "v"(lds_off),
                  "v"((unsigned long long)(uintptr_t)gsrc)
               : "memory");
}
__device__ __forceinline__ void wait_asynccnt0() {
  asm volatile("s_wait_asynccnt 0x0" ::: "memory");
}

// ---- TDM: 2D tile (tile_d1 rows x tile_d0 elems, 2-byte elems) -> LDS ------
// D# per CDNA5 ISA. Groups 2/3 zero (2D tensor).
__device__ __forceinline__ void tdm_load_2d_bf16(const void* gsrc,
                                                 unsigned int lds_addr,
                                                 unsigned int tile_d0,
                                                 unsigned int tile_d1,
                                                 unsigned int tensor_d0,
                                                 unsigned int tensor_d1,
                                                 unsigned long long stride0) {
#if HAVE_TDM
  unsigned long long ga = (unsigned long long)(uintptr_t)gsrc;
  uint32x4 g0;
  g0.x = 1u;                                           // count=1, user mode
  g0.y = lds_addr;                                     // lds_addr [63:32]
  g0.z = (unsigned int)(ga & 0xffffffffu);             // global_addr[31:0]
  g0.w = (unsigned int)((ga >> 32) & 0x1ffffffu)       // global_addr[56:32]
         | (2u << 30);                                 // type=2 ("image")
  int32x8 g1;
  g1[0] = (int)(1u << 16);                             // data_size=1 (2B)
  g1[1] = (int)((tensor_d0 & 0xffffu) << 16);          // tensor_dim0[15:0]
  g1[2] = (int)((tensor_d0 >> 16) | ((tensor_d1 & 0xffffu) << 16));
  g1[3] = (int)((tensor_d1 >> 16) | (tile_d0 << 16));  // tile_dim0
  g1[4] = (int)(tile_d1 & 0xffffu);                    // tile_dim1 (dim2=0)
  g1[5] = (int)(unsigned int)(stride0 & 0xffffffffu);  // dim0_stride[31:0]
  g1[6] = (int)(unsigned int)((stride0 >> 32) & 0xffffu);
  g1[7] = 0;
  int32x4 z4 = {0, 0, 0, 0};
#if __clang_major__ >= 23
  int32x8 z8 = {0, 0, 0, 0, 0, 0, 0, 0};
  __builtin_amdgcn_tensor_load_to_lds(g0, g1, z4, z4, z8, 0);
#else
  __builtin_amdgcn_tensor_load_to_lds(g0, g1, z4, z4, 0);
#endif
#endif
}

// --------------------------- cast kernel -----------------------------------
__global__ __launch_bounds__(256) void cast_f32_bf16(const float* __restrict__ in,
                                                     __bf16* __restrict__ out,
                                                     int n) {
  int i = (blockIdx.x * 256 + threadIdx.x) * 4;
#pragma unroll
  for (int j = 0; j < 4; ++j)
    if (i + j < n) out[i + j] = (__bf16)in[i + j];
}

// --------------------------- NT GEMM ---------------------------------------
// C[M,N] = A[M,K] * B[N,K]^T ; A,B bf16 row-major (K contiguous), f32 accum.
// FINAL=false: store bf16. FINAL=true: store f32 + bias[n].
#define BM 128
#define BN 64
#define BK 32

template <bool FINAL>
__global__ __launch_bounds__(256) void gemm_nt(const __bf16* __restrict__ A,
                                               const __bf16* __restrict__ Bw,
                                               void* __restrict__ Cout,
                                               int M, int N, int K,
                                               const float* __restrict__ bias) {
  __shared__ alignas(32) __bf16 sA[BM * BK];
  __shared__ alignas(32) __bf16 sB[BN * BK];

  const int tid  = threadIdx.x;
  const int wave = tid >> 5;
  const int lane = tid & 31;
  const int half = lane >> 4;
  const int l15  = lane & 15;

  const int nTilesN = N / BN;
  const int tileM = (blockIdx.x / nTilesN) * BM;
  const int tileN = (blockIdx.x % nTilesN) * BN;

  // 8 waves: 4 (M) x 2 (N); each wave owns a 32x32 sub-tile (2x2 WMMA tiles).
  const int wm = (wave >> 1) * 32;
  const int wn = (wave & 1) * 32;

  v8f acc[2][2] = {};

  for (int k0 = 0; k0 < K; k0 += BK) {
    // ---- Stage A tile (128x32): 512 x 16B async copies, ASYNCcnt-tracked.
#pragma unroll
    for (int i = 0; i < 2; ++i) {
      int idx = tid + i * 256;
      int row = idx >> 2;
      int cg  = (idx & 3) << 3;
      async_copy_b128((unsigned int)(uintptr_t)&sA[row * BK + cg],
                      &A[(size_t)(tileM + row) * K + k0 + cg]);
    }
    // ---- Stage B tile (64x32) via the Tensor Data Mover (one D# issue).
#if HAVE_TDM
    if (tid == 0) {
      tdm_load_2d_bf16(&Bw[(size_t)tileN * K + k0],
                       (unsigned int)(uintptr_t)&sB[0],
                       /*tile_d0=*/BK, /*tile_d1=*/BN,
                       /*tensor_d0=*/(unsigned int)K,
                       /*tensor_d1=*/(unsigned int)N,
                       /*stride0=*/(unsigned long long)K);
      __builtin_amdgcn_s_wait_tensorcnt(0);
    }
#else
    {
      int row = tid >> 2;
      int cg  = (tid & 3) << 3;
      *(v8bf*)&sB[row * BK + cg] =
          *(const v8bf*)&Bw[(size_t)(tileN + row) * K + k0 + cg];
    }
#endif
    if (k0 + BK < K) {
      __builtin_prefetch(&A[(size_t)(tileM + (tid >> 2)) * K + k0 + BK], 0, 0);
      __builtin_prefetch(&Bw[(size_t)(tileN + (tid >> 2)) * K + k0 + BK], 0, 0);
    }
    wait_asynccnt0();
    __syncthreads();

    v16bf afrag[2], bfrag[2];
#pragma unroll
    for (int i = 0; i < 2; ++i) {
      const __bf16* ar = &sA[(wm + i * 16 + l15) * BK + half * 16];
      afrag[i] = load_frag2(ar, ar + 8);
    }
#pragma unroll
    for (int j = 0; j < 2; ++j) {
      const __bf16* br = &sB[(wn + j * 16 + l15) * BK + half * 16];
      bfrag[j] = *(const v16bf*)br;   // 16 contiguous K values, 32B aligned
    }
#pragma unroll
    for (int i = 0; i < 2; ++i)
#pragma unroll
      for (int j = 0; j < 2; ++j)
        acc[i][j] = wmma_bf16(afrag[i], bfrag[j], acc[i][j]);
    __syncthreads();
  }

  // Epilogue. C layout: vgpr r -> m = r + 8*half ; n = lane&15.
#pragma unroll
  for (int i = 0; i < 2; ++i)
#pragma unroll
    for (int j = 0; j < 2; ++j) {
      int mrow = tileM + wm + i * 16 + half * 8;
      int ncol = tileN + wn + j * 16 + l15;
#pragma unroll
      for (int r = 0; r < 8; ++r) {
        float v = acc[i][j][r];
        if (FINAL) {
          ((float*)Cout)[(size_t)(mrow + r) * N + ncol] = v + bias[ncol];
        } else {
          ((__bf16*)Cout)[(size_t)(mrow + r) * N + ncol] = (__bf16)v;
        }
      }
    }
}

// ----------------------- per-head V transpose ------------------------------
// vt[(b*8+h)*96 + d][kv] = vp[b*1024 + kv][h*96 + d]
__global__ __launch_bounds__(256) void transpose_v(const __bf16* __restrict__ vp,
                                                   __bf16* __restrict__ vt) {
  size_t i = (size_t)blockIdx.x * 256 + threadIdx.x;   // output index
  int kv = (int)(i & 1023);
  size_t r = i >> 10;          // bh*96 + d
  int d  = (int)(r % 96);
  int bh = (int)(r / 96);
  int b = bh >> 3, h = bh & 7;
  vt[i] = vp[((size_t)b * 1024 + kv) * DIM + h * HDIM + d];
}

// --------------------------- flash attention -------------------------------
// grid.x = B*H*(NQ/64); block = 128 (4 waves); each wave owns 16 Q rows.
__global__ __launch_bounds__(128) void attn_kernel(const __bf16* __restrict__ qp,
                                                   const __bf16* __restrict__ kp,
                                                   const __bf16* __restrict__ vt,
                                                   __bf16* __restrict__ xo) {
  const int blk   = blockIdx.x;
  const int qtile = blk & 15;
  const int h     = (blk >> 4) & 7;
  const int b     = blk >> 7;
  const int wave  = threadIdx.x >> 5;
  const int lane  = threadIdx.x & 31;
  const int half  = lane >> 4;
  const int l15   = lane & 15;
  const int qrow0 = qtile * 64 + wave * 16;

  __shared__ alignas(32) __bf16 sP[4][16 * 64];   // per-wave P tile (bf16)

  const float scale = 1.0f / (float)HDIM;

  // Q fragments (held in registers across the whole KV sweep).
  const __bf16* qrow = qp + ((size_t)(b * NQ) + qrow0 + l15) * DIM + h * HDIM;
  v16bf qf[3];
#pragma unroll
  for (int ks = 0; ks < 3; ++ks) {
    const __bf16* p = qrow + ks * 32 + half * 16;
    qf[ks] = load_frag2(p, p + 8);
  }

  float mrun[8], lrun[8];
  v8f o[6] = {};
#pragma unroll
  for (int r = 0; r < 8; ++r) { mrun[r] = -1e30f; lrun[r] = 0.0f; }

  for (int kv0 = 0; kv0 < NK; kv0 += 64) {
    // ---- S = Q·K^T (16x64, 4 n-tiles x 3 k-steps) ----
    v8f s[4] = {};
#pragma unroll
    for (int nt = 0; nt < 4; ++nt) {
      const __bf16* krow =
          kp + ((size_t)(b * NK) + kv0 + nt * 16 + l15) * DIM + h * HDIM;
#pragma unroll
      for (int ks = 0; ks < 3; ++ks) {
        v16bf bf = *(const v16bf*)(krow + ks * 32 + half * 16);
        s[nt] = wmma_bf16(qf[ks], bf, s[nt]);
      }
    }
    // ---- online softmax (row = r + 8*half; columns live across 16 lanes) ----
#pragma unroll
    for (int r = 0; r < 8; ++r) {
      float mx = -1e30f;
#pragma unroll
      for (int nt = 0; nt < 4; ++nt) {
        s[nt][r] = s[nt][r] * scale;
        mx = fmaxf(mx, s[nt][r]);
      }
#pragma unroll
      for (int m = 1; m < 16; m <<= 1) mx = fmaxf(mx, __shfl_xor(mx, m, 32));
      float nm    = fmaxf(mrun[r], mx);
      float alpha = __expf(mrun[r] - nm);
      mrun[r] = nm;
      float rsum = 0.0f;
#pragma unroll
      for (int nt = 0; nt < 4; ++nt) {
        float p = __expf(s[nt][r] - nm);
        s[nt][r] = p;
        rsum += p;
      }
#pragma unroll
      for (int m = 1; m < 16; m <<= 1) rsum += __shfl_xor(rsum, m, 32);
      lrun[r] = lrun[r] * alpha + rsum;
#pragma unroll
      for (int d = 0; d < 6; ++d) o[d][r] = o[d][r] * alpha;
    }
    // ---- P -> LDS (bf16, [16][64]) ; per-wave DS ops are in-order ----
    __bf16* pb = &sP[wave][0];
#pragma unroll
    for (int nt = 0; nt < 4; ++nt)
#pragma unroll
      for (int r = 0; r < 8; ++r)
        pb[(r + half * 8) * 64 + nt * 16 + l15] = (__bf16)s[nt][r];
    // Re-fragment P as A-matrix tiles (2 k-steps over the 64 kv columns).
    v16bf pf[2];
#pragma unroll
    for (int ks = 0; ks < 2; ++ks) {
      const __bf16* pr = pb + l15 * 64 + ks * 32 + half * 16;
      pf[ks] = load_frag2(pr, pr + 8);
    }
    // ---- O += P·V  (V^T rows are kv-contiguous) ----
    const __bf16* vbase = vt + ((size_t)((b * 8 + h) * HDIM)) * NK + kv0;
#pragma unroll
    for (int d = 0; d < 6; ++d) {
      const __bf16* vrow = vbase + (size_t)(d * 16 + l15) * NK;
#pragma unroll
      for (int ks = 0; ks < 2; ++ks) {
        v16bf bf = *(const v16bf*)(vrow + ks * 32 + half * 16);
        o[d] = wmma_bf16(pf[ks], bf, o[d]);
      }
    }
  }

  // ---- epilogue: x[b, q, h*96 + d] = O / l ----
#pragma unroll
  for (int d = 0; d < 6; ++d)
#pragma unroll
    for (int r = 0; r < 8; ++r) {
      float val = o[d][r] / lrun[r];
      int q   = qrow0 + r + half * 8;
      int col = h * HDIM + d * 16 + l15;
      xo[((size_t)(b * NQ) + q) * DIM + col] = (__bf16)val;
    }
}

// --------------------------- launch ----------------------------------------
extern "C" void kernel_launch(void* const* d_in, const int* in_sizes, int n_in,
                              void* d_out, int out_size, void* d_ws, size_t ws_size,
                              hipStream_t stream) {
  const float* q  = (const float*)d_in[0];
  const float* kv = (const float*)d_in[1];
  const float* wq = (const float*)d_in[2];
  const float* wk = (const float*)d_in[3];
  const float* wv = (const float*)d_in[4];
  const float* wp = (const float*)d_in[5];
  const float* bp = (const float*)d_in[6];

  char* ws = (char*)d_ws;
  size_t off = 0;
  auto alloc = [&](size_t elems) {
    __bf16* p = (__bf16*)(ws + off);
    off += elems * sizeof(__bf16);
    return p;
  };
  const size_t nQ  = (size_t)MROWS * DIM;       // 6291456
  const size_t nKV = (size_t)MROWS * DIM_EDGE;  // 2097152
  __bf16* qb  = alloc(nQ);
  __bf16* kvb = alloc(nKV);
  __bf16* wqb = alloc((size_t)DIM * DIM);
  __bf16* wkb = alloc((size_t)DIM * DIM_EDGE);
  __bf16* wvb = alloc((size_t)DIM * DIM_EDGE);
  __bf16* wpb = alloc((size_t)DIM * DIM);
  __bf16* qp  = alloc(nQ);
  __bf16* kp  = alloc(nQ);
  __bf16* vp  = alloc(nQ);
  __bf16* vt  = alloc(nQ);
  __bf16* xo  = alloc(nQ);

  struct { const float* src; __bf16* dst; int n; } casts[6] = {
      {q,  qb,  (int)nQ},            {kv, kvb, (int)nKV},
      {wq, wqb, DIM * DIM},          {wk, wkb, DIM * DIM_EDGE},
      {wv, wvb, DIM * DIM_EDGE},     {wp, wpb, DIM * DIM}};
  for (int i = 0; i < 6; ++i) {
    int blocks = (casts[i].n + 1023) / 1024;
    cast_f32_bf16<<<blocks, 256, 0, stream>>>(casts[i].src, casts[i].dst,
                                              casts[i].n);
  }

  const int gridProj = (MROWS / BM) * (DIM / BN);   // 64*12 = 768
  gemm_nt<false><<<gridProj, 256, 0, stream>>>(qb,  wqb, qp, MROWS, DIM, DIM,      nullptr);
  gemm_nt<false><<<gridProj, 256, 0, stream>>>(kvb, wkb, kp, MROWS, DIM, DIM_EDGE, nullptr);
  gemm_nt<false><<<gridProj, 256, 0, stream>>>(kvb, wvb, vp, MROWS, DIM, DIM_EDGE, nullptr);

  transpose_v<<<(int)(nQ / 256), 256, 0, stream>>>(vp, vt);

  attn_kernel<<<BATCH * NHEAD * (NQ / 64), 128, 0, stream>>>(qp, kp, vt, xo);

  gemm_nt<true><<<gridProj, 256, 0, stream>>>(xo, wpb, d_out, MROWS, DIM, DIM, bp);
}
